// InternVL3VisionLayer_12360915878694
// MI455X (gfx1250) — matrix operand order repaired and checked by reading, not compiled, not run on hardware
//
#include <hip/hip_runtime.h>
#include <cstdint>
#include <cstddef>

// ---------------------------------------------------------------------------
// MI455X (gfx1250) ViT encoder layer: bf16 WMMA GEMMs (async-LDS double
// buffered) + WMMA flash attention.
// ---------------------------------------------------------------------------

typedef __bf16 bf16_t;
typedef __attribute__((ext_vector_type(16))) __bf16 v16bf;
typedef __attribute__((ext_vector_type(8)))  __bf16 v8bf;
typedef __attribute__((ext_vector_type(4)))  __bf16 v4bf;
typedef __attribute__((ext_vector_type(8)))  float  v8f;
typedef __attribute__((ext_vector_type(4)))  float  v4f;
typedef __attribute__((ext_vector_type(4)))  int    v4i;

#define GLOBAL_AS __attribute__((address_space(1)))
#define LDS_AS    __attribute__((address_space(3)))

#if defined(__has_builtin)
#if __has_builtin(__builtin_amdgcn_global_load_async_to_lds_b128) && \
    __has_builtin(__builtin_amdgcn_s_wait_asynccnt)
#define USE_ASYNC_LDS 1
#endif
#endif

static __device__ __forceinline__ int imin(int a, int b) { return a < b ? a : b; }

static __device__ __forceinline__ v16bf cat8(v8bf lo, v8bf hi) {
  return __builtin_shufflevector(lo, hi, 0,1,2,3,4,5,6,7,8,9,10,11,12,13,14,15);
}

static __device__ __forceinline__ v8f wmma_bf16(v16bf a, v16bf b, v8f c) {
  // D(f32 16x16) = A(bf16 16x32) * B(bf16 32x16) + C
  return __builtin_amdgcn_wmma_f32_16x16x32_bf16(false, a, false, b, (short)0, c,
                                                 false, false);
}

// 16B global -> LDS copy; async (ASYNCcnt-tracked) when the toolchain has it.
static __device__ __forceinline__ void copy16(const bf16_t* g, bf16_t* l) {
#ifdef USE_ASYNC_LDS
  __builtin_amdgcn_global_load_async_to_lds_b128(
      (GLOBAL_AS v4i*)(uintptr_t)g, (LDS_AS v4i*)(uintptr_t)l, 0, 0);
#else
  *(v8bf*)l = *(const v8bf*)g;
#endif
}
static __device__ __forceinline__ void wait_async(int n) {
#ifdef USE_ASYNC_LDS
  if (n == 0) __builtin_amdgcn_s_wait_asynccnt(0);
  else        __builtin_amdgcn_s_wait_asynccnt(3);
#else
  (void)n;
#endif
}

// ---------------------------------------------------------------------------
// Weight transpose + convert: W[N x K] f32 -> Wt[K x N] bf16 (LDS tiled).
// ---------------------------------------------------------------------------
__global__ __launch_bounds__(256)
void wT_bf16(const float* __restrict__ W, bf16_t* __restrict__ Wt, int N, int K) {
  __shared__ bf16_t t[32][33];
  const int k0 = blockIdx.x * 32, n0 = blockIdx.y * 32;
  const int tx = threadIdx.x, ty = threadIdx.y;  // (32, 8)
  #pragma unroll
  for (int p = 0; p < 4; ++p)
    t[ty + p * 8][tx] = (bf16_t)W[(size_t)(n0 + ty + p * 8) * K + k0 + tx];
  __syncthreads();
  #pragma unroll
  for (int p = 0; p < 4; ++p)
    Wt[(size_t)(k0 + ty + p * 8) * N + n0 + tx] = t[tx][ty + p * 8];
}

// ---------------------------------------------------------------------------
// LayerNorm (H=1024) -> bf16.  One 256-thread block (8 waves) per token.
// ---------------------------------------------------------------------------
__global__ __launch_bounds__(256)
void ln_to_bf16(const float* __restrict__ x, const float* __restrict__ w,
                const float* __restrict__ b, bf16_t* __restrict__ out) {
  const int m = blockIdx.x;
  const int tid = threadIdx.x;
  const float* xp = x + (size_t)m * 1024 + tid * 4;
  v4f v = *(const v4f*)xp;
  float s  = v.x + v.y + v.z + v.w;
  float s2 = v.x*v.x + v.y*v.y + v.z*v.z + v.w*v.w;
  #pragma unroll
  for (int off = 16; off >= 1; off >>= 1) {
    s  += __shfl_xor(s,  off, 32);
    s2 += __shfl_xor(s2, off, 32);
  }
  __shared__ float red[2][8];
  if ((tid & 31) == 0) { red[0][tid >> 5] = s; red[1][tid >> 5] = s2; }
  __syncthreads();
  float ts = 0.f, ts2 = 0.f;
  #pragma unroll
  for (int i = 0; i < 8; ++i) { ts += red[0][i]; ts2 += red[1][i]; }
  const float mu  = ts  * (1.f / 1024.f);
  const float var = ts2 * (1.f / 1024.f) - mu * mu;
  const float rs  = rsqrtf(var + 1e-6f);
  const float* wp = w + tid * 4;
  const float* bp = b + tid * 4;
  v4bf o;
  #pragma unroll
  for (int j = 0; j < 4; ++j) o[j] = (bf16_t)((v[j] - mu) * rs * wp[j] + bp[j]);
  *(v4bf*)(out + (size_t)m * 1024 + tid * 4) = o;
}

// ---------------------------------------------------------------------------
// Tiled WMMA GEMM:  out[M x N] = A[M x K](bf16) @ Wt[K x N](bf16)
// Block tile 64x128x32, 8 waves (2x4), each wave 32x32 (4 WMMA accumulators).
// Double-buffered LDS staging via async global->LDS loads when available.
// MODE 0: bf16 out = acc + bias                       (QKV)
// MODE 1: bf16 out = gelu(acc + bias), exact erf      (fc1)
// MODE 2: f32  out = resid + ls * (acc + bias)        (proj / fc2)
// ---------------------------------------------------------------------------
template <int MODE>
__global__ __launch_bounds__(256)
void gemm_bf16(const bf16_t* __restrict__ A, const bf16_t* __restrict__ Bt,
               const float* __restrict__ bias, const float* __restrict__ resid,
               const float* __restrict__ ls, bf16_t* __restrict__ outb,
               float* __restrict__ outf, int M, int N, int K, int ldo) {
  constexpr int BM = 64, BN = 128, BK = 32;
  __shared__ bf16_t sA[2][BM][BK + 8];   // row stride 80B (16B multiple)
  __shared__ bf16_t sB[2][BK][BN + 8];   // row stride 272B (16B multiple)

  const int tid  = threadIdx.x;
  const int lane = tid & 31;
  const int wave = tid >> 5;   // 0..7
  const int wm   = wave >> 2;  // 0..1  (M offset 32*wm)
  const int wn   = wave & 3;   // 0..3  (N offset 32*wn)
  const int m0   = blockIdx.x * BM;
  const int n0   = blockIdx.y * BN;

  // staging coordinates (3 x 16B per thread per K-tile)
  const int ra = tid >> 2, ca = (tid & 3) * 8;       // A: 64 rows x 32
  const int ga = imin(m0 + ra, M - 1);               // clamp (rows >= M unused)
  const int kb = tid >> 4, cb = (tid & 15) * 8;      // B: 32 rows x 128

  auto stage = [&](int k0, int buf) {
    copy16(A  + (size_t)ga * K + k0 + ca,             &sA[buf][ra][ca]);
    copy16(Bt + (size_t)(k0 + kb) * N + n0 + cb,      &sB[buf][kb][cb]);
    copy16(Bt + (size_t)(k0 + kb + 16) * N + n0 + cb, &sB[buf][kb + 16][cb]);
  };

  v8f acc[2][2] = {};

  stage(0, 0);
  int buf = 0;
  for (int k0 = 0; k0 < K; k0 += BK) {
    const bool has_next = (k0 + BK) < K;
    if (has_next) stage(k0 + BK, buf ^ 1);
    wait_async(has_next ? 3 : 0);  // in-order: oldest 3 (tile k0) have landed
    __syncthreads();

    // ---- fragments (ISA 7.12.2 layouts) ----
    v16bf af[2], bfg[2];
    {
      const int r  = lane & 15;
      const int kf = (lane >> 4) * 8;
      #pragma unroll
      for (int t = 0; t < 2; ++t) {
        const bf16_t* p = &sA[buf][wm * 32 + t * 16 + r][0];
        af[t] = cat8(*(const v8bf*)(p + kf), *(const v8bf*)(p + kf + 16));
      }
      #pragma unroll
      for (int t = 0; t < 2; ++t) {
        const int n = wn * 32 + t * 16 + r;
        #pragma unroll
        for (int u = 0; u < 8; ++u) {
          bfg[t][u]     = sB[buf][kf + u][n];
          bfg[t][u + 8] = sB[buf][kf + 16 + u][n];
        }
      }
    }
    #pragma unroll
    for (int mt = 0; mt < 2; ++mt)
      #pragma unroll
      for (int nt = 0; nt < 2; ++nt)
        acc[mt][nt] = wmma_bf16(af[mt], bfg[nt], acc[mt][nt]);
    __syncthreads();
    buf ^= 1;
  }

  // ---- epilogue (C/D layout: VGPR i -> row i / i+8; lane&15 -> col) ----
  const int cit  = lane & 15;
  const int rsel = (lane >> 4) * 8;
  #pragma unroll
  for (int mt = 0; mt < 2; ++mt) {
    #pragma unroll
    for (int nt = 0; nt < 2; ++nt) {
      const int n = n0 + wn * 32 + nt * 16 + cit;
      const float bn = bias[n];
      float lsn = 0.f;
      if constexpr (MODE == 2) lsn = ls[n];
      #pragma unroll
      for (int i = 0; i < 8; ++i) {
        const int m = m0 + wm * 32 + mt * 16 + i + rsel;
        if (m < M) {
          const float v = acc[mt][nt][i] + bn;
          if constexpr (MODE == 0) {
            outb[(size_t)m * ldo + n] = (bf16_t)v;
          } else if constexpr (MODE == 1) {
            const float g = 0.5f * v * (1.f + erff(v * 0.70710678118654752f));
            outb[(size_t)m * ldo + n] = (bf16_t)g;
          } else {
            outf[(size_t)m * ldo + n] = resid[(size_t)m * ldo + n] + lsn * v;
          }
        }
      }
    }
  }
}

// ---------------------------------------------------------------------------
// Flash attention: one wave per (b, h, 16-query tile).  N=1025, HD=64.
// qkv: bf16 [B*1025 x 3072]; cols 0..1023 = Q, 1024..2047 = K, 2048..3071 = V
// K is staged TRANSPOSED in LDS; B-fragments are read one bf16 per Kt row (strided).
// ---------------------------------------------------------------------------
__global__ __launch_bounds__(32)
void attn_flash(const bf16_t* __restrict__ qkv, bf16_t* __restrict__ out) {
  const int NTOK = 1025, LDQ = 3072;
  const int b    = blockIdx.y >> 4;
  const int h    = blockIdx.y & 15;
  const int q0   = blockIdx.x * 16;
  const int lane = threadIdx.x;

  const size_t base = (size_t)b * NTOK * LDQ;
  const bf16_t* Qp = qkv + base + h * 64;
  const bf16_t* Kp = qkv + base + 1024 + h * 64;
  const bf16_t* Vp = qkv + base + 2048 + h * 64;

  __shared__ bf16_t Kt[64][40];  // [d][key]   row stride 80B (16B mult)
  __shared__ bf16_t Pl[16][40];  // [qrow][key] row stride 80B (16B mult)

  // Q A-fragments for d-tiles {0..31},{32..63}
  v16bf aQ[2];
  {
    const int r  = lane & 15;
    const int kf = (lane >> 4) * 8;
    const int t  = imin(q0 + r, NTOK - 1);
    const bf16_t* qp = Qp + (size_t)t * LDQ;
    #pragma unroll
    for (int dt = 0; dt < 2; ++dt)
      aQ[dt] = cat8(*(const v8bf*)(qp + dt * 32 + kf),
                    *(const v8bf*)(qp + dt * 32 + kf + 16));
  }

  float mrow[8], lrow[8];
  v8f o[4] = {};
  #pragma unroll
  for (int i = 0; i < 8; ++i) { mrow[i] = -3.0e38f; lrow[i] = 0.f; }
  const float scale = 0.125f;  // 64^-0.5

  for (int kc = 0; kc < NTOK; kc += 32) {
    // ---- stage 32-key K chunk transposed: Kt[d][j] = K[kc+j][d] ----
    {
      const int j = imin(kc + lane, NTOK - 1);
      const bf16_t* kp = Kp + (size_t)j * LDQ;
      #pragma unroll
      for (int c = 0; c < 8; ++c) {
        v8bf v = *(const v8bf*)(kp + c * 8);
        #pragma unroll
        for (int u = 0; u < 8; ++u) Kt[c * 8 + u][lane] = v[u];
      }
    }
    __syncthreads();

    // ---- S = Q @ K^T : B-fragments gathered element-wise down Kt columns ----
    v8f s0 = {}, s1 = {};
    #pragma unroll
    for (int dt = 0; dt < 2; ++dt) {
      const int kf = (lane >> 4) * 8, jn = lane & 15;   // B col = lane&15 (key)
      v16bf b0, b1;
      #pragma unroll
      for (int u = 0; u < 8; ++u) {
        b0[u]     = Kt[dt * 32 + kf + u][jn];
        b0[u + 8] = Kt[dt * 32 + kf + 16 + u][jn];
        b1[u]     = Kt[dt * 32 + kf + u][16 + jn];
        b1[u + 8] = Kt[dt * 32 + kf + 16 + u][16 + jn];
      }
      s0 = wmma_bf16(aQ[dt], b0, s0);
      s1 = wmma_bf16(aQ[dt], b1, s1);
    }

    // ---- mask + online softmax (state per accumulator slot / row) ----
    const int  j0  = kc + (lane & 15);
    const bool ok0 = (j0 < NTOK), ok1 = (j0 + 16 < NTOK);
    #pragma unroll
    for (int i = 0; i < 8; ++i) {
      s0[i] = ok0 ? s0[i] * scale : -3.0e38f;
      s1[i] = ok1 ? s1[i] * scale : -3.0e38f;
      float v = fmaxf(s0[i], s1[i]);
      v = fmaxf(v, __shfl_xor(v, 1, 32));
      v = fmaxf(v, __shfl_xor(v, 2, 32));
      v = fmaxf(v, __shfl_xor(v, 4, 32));
      v = fmaxf(v, __shfl_xor(v, 8, 32));
      const float mn   = fmaxf(mrow[i], v);
      const float corr = __expf(mrow[i] - mn);
      mrow[i] = mn;
      const float e0 = __expf(s0[i] - mn);
      const float e1 = __expf(s1[i] - mn);
      s0[i] = e0; s1[i] = e1;
      float rs = e0 + e1;
      rs += __shfl_xor(rs, 1, 32);
      rs += __shfl_xor(rs, 2, 32);
      rs += __shfl_xor(rs, 4, 32);
      rs += __shfl_xor(rs, 8, 32);
      lrow[i] = lrow[i] * corr + rs;
      #pragma unroll
      for (int dt = 0; dt < 4; ++dt) o[dt][i] *= corr;
    }

    // ---- P: C-layout -> LDS -> A-layout ----
    __syncthreads();
    {
      const int jn = lane & 15;
      const int rsel = (lane >> 4) * 8;
      #pragma unroll
      for (int i = 0; i < 8; ++i) {
        Pl[i + rsel][jn]      = (bf16_t)s0[i];
        Pl[i + rsel][16 + jn] = (bf16_t)s1[i];
      }
    }
    __syncthreads();
    v16bf aP;
    {
      const int r  = lane & 15;
      const int kf = (lane >> 4) * 8;
      aP = cat8(*(const v8bf*)(&Pl[r][kf]), *(const v8bf*)(&Pl[r][kf + 16]));
    }

    // ---- O += P @ V  (V B-fragments gathered from global, col = lane&15) ----
    const int kfv = (lane >> 4) * 8, dn = lane & 15;
    #pragma unroll
    for (int dt = 0; dt < 4; ++dt) {
      v16bf bv;
      #pragma unroll
      for (int u = 0; u < 8; ++u) {
        const int ja = imin(kc + kfv + u, NTOK - 1);
        const int jb = imin(kc + kfv + 16 + u, NTOK - 1);
        bv[u]     = Vp[(size_t)ja * LDQ + dt * 16 + dn];
        bv[u + 8] = Vp[(size_t)jb * LDQ + dt * 16 + dn];
      }
      o[dt] = wmma_bf16(aP, bv, o[dt]);
    }
    __syncthreads();
  }

  // ---- normalize + store ----
  {
    const int jn = lane & 15;
    const int rsel = (lane >> 4) * 8;
    #pragma unroll
    for (int i = 0; i < 8; ++i) {
      const int t = q0 + i + rsel;
      if (t < NTOK) {
        const float inv = 1.f / fmaxf(lrow[i], 1e-20f);
        #pragma unroll
        for (int dt = 0; dt < 4; ++dt)
          out[((size_t)b * NTOK + t) * 1024 + h * 64 + dt * 16 + jn] =
              (bf16_t)(o[dt][i] * inv);
      }
    }
  }
}

// ---------------------------------------------------------------------------
extern "C" void kernel_launch(void* const* d_in, const int* in_sizes, int n_in,
                              void* d_out, int out_size, void* d_ws,
                              size_t ws_size, hipStream_t stream) {
  (void)in_sizes; (void)n_in; (void)out_size; (void)ws_size;
  const float* x      = (const float*)d_in[0];
  const float* qkv_w  = (const float*)d_in[1];
  const float* qkv_b  = (const float*)d_in[2];
  const float* proj_w = (const float*)d_in[3];
  const float* proj_b = (const float*)d_in[4];
  const float* fc1_w  = (const float*)d_in[5];
  const float* fc1_b  = (const float*)d_in[6];
  const float* fc2_w  = (const float*)d_in[7];
  const float* fc2_b  = (const float*)d_in[8];
  const float* n1w    = (const float*)d_in[9];
  const float* n1b    = (const float*)d_in[10];
  const float* n2w    = (const float*)d_in[11];
  const float* n2b    = (const float*)d_in[12];
  const float* ls1    = (const float*)d_in[13];
  const float* ls2    = (const float*)d_in[14];
  float* out = (float*)d_out;

  const int M = 8 * 1025;  // 8200 tokens
  bf16_t* ws      = (bf16_t*)d_ws;
  bf16_t* xn      = ws;                        // M*1024
  bf16_t* qkv     = ws + (size_t)M * 1024;     // M*3072
  bf16_t* attn_o  = qkv + (size_t)M * 3072;    // M*1024
  bf16_t* hbuf    = qkv;                       // M*4096 (aliases qkv+attn_o)
  bf16_t* wt_qkv  = ws + (size_t)M * 5120;     // [1024 x 3072]
  bf16_t* wt_proj = wt_qkv + (size_t)3072 * 1024;  // [1024 x 1024]
  bf16_t* wt_fc1  = wt_proj + (size_t)1024 * 1024; // [1024 x 4096]
  bf16_t* wt_fc2  = wt_fc1 + (size_t)4096 * 1024;  // [4096 x 1024]

  const dim3 b256(256), bT(32, 8);
  const int MB = (M + 63) / 64;  // 129 M-tiles

  // one-time weight transpose+convert to bf16 [K x N]
  wT_bf16<<<dim3(1024 / 32, 3072 / 32), bT, 0, stream>>>(qkv_w, wt_qkv, 3072, 1024);
  wT_bf16<<<dim3(1024 / 32, 1024 / 32), bT, 0, stream>>>(proj_w, wt_proj, 1024, 1024);
  wT_bf16<<<dim3(1024 / 32, 4096 / 32), bT, 0, stream>>>(fc1_w, wt_fc1, 4096, 1024);
  wT_bf16<<<dim3(4096 / 32, 1024 / 32), bT, 0, stream>>>(fc2_w, wt_fc2, 1024, 4096);

  // x1 = x + ls1 * attn(LN1(x))
  ln_to_bf16<<<M, b256, 0, stream>>>(x, n1w, n1b, xn);
  gemm_bf16<0><<<dim3(MB, 3072 / 128), b256, 0, stream>>>(
      xn, wt_qkv, qkv_b, nullptr, nullptr, qkv, nullptr, M, 3072, 1024, 3072);
  attn_flash<<<dim3(65, 128), dim3(32), 0, stream>>>(qkv, attn_o);
  gemm_bf16<2><<<dim3(MB, 1024 / 128), b256, 0, stream>>>(
      attn_o, wt_proj, proj_b, x, ls1, nullptr, out, M, 1024, 1024, 1024);

  // out = x1 + ls2 * mlp(LN2(x1))
  ln_to_bf16<<<M, b256, 0, stream>>>(out, n2w, n2b, xn);
  gemm_bf16<1><<<dim3(MB, 4096 / 128), b256, 0, stream>>>(
      xn, wt_fc1, fc1_b, nullptr, nullptr, hbuf, nullptr, M, 4096, 1024, 4096);
  gemm_bf16<2><<<dim3(MB, 1024 / 128), b256, 0, stream>>>(
      hbuf, wt_fc2, fc2_b, out, ls2, nullptr, out, M, 1024, 4096, 1024);
}
